// Cross_Attention_64029372448774
// MI455X (gfx1250) — compile-verified, matrix operand order checked by reading, not run on hardware
//
#include <hip/hip_runtime.h>
#include <stdint.h>

#define EMBED 1024
#define NHEAD 16
#define DK    64
#define BATCH 4
#define SDEC  1024
#define SENC  2048

typedef __attribute__((ext_vector_type(16))) __bf16 v16bf;
typedef __attribute__((ext_vector_type(8)))  __bf16 v8bf;
typedef __attribute__((ext_vector_type(8)))  float  v8f;

static __device__ __forceinline__ v8f wmma_bf16(v16bf a, v16bf b, v8f c) {
  // (neg_a, A, neg_b, B, c_mod, C, reuse_a, reuse_b)
  return __builtin_amdgcn_wmma_f32_16x16x32_bf16(false, a, false, b, (short)0, c,
                                                 false, false);
}

// Load the two 16-byte chunks of a 16-bit WMMA operand for this lane.
static __device__ __forceinline__ v16bf ld2x8_bf16(const __bf16* p0, const __bf16* p1) {
  union { v16bf v; v8bf h[2]; } u;
  u.h[0] = *(const v8bf*)p0;
  u.h[1] = *(const v8bf*)p1;
  return u.v;
}

// ---------------------------------------------------------------------------
// Stage 1: W (f32, [k][n]) -> Wt (bf16, [n][k]) so B-operand rows are contiguous
// ---------------------------------------------------------------------------
__global__ __launch_bounds__(256) void wt_bf16_kernel(const float* __restrict__ W,
                                                      __bf16* __restrict__ Wt) {
  const int idx = blockIdx.x * 256 + threadIdx.x;   // 1M threads, coalesced read
  const int k = idx >> 10;
  const int n = idx & 1023;
  Wt[(size_t)n * EMBED + k] = (__bf16)W[idx];
}

// ---------------------------------------------------------------------------
// Tiled WMMA GEMM, Y = A @ W + bias, with block-shared W tile in LDS.
//   AMODE: 0 = A is f32 (convert on load), 1 = A is bf16
//   SMODE: 0 = bf16 row-major out, 1 = bf16 V-transposed out [b][h][d][s]
//          (packed b128 stores), 2 = f32 row-major out
// Block = 8 waves stacked along M (128 rows), all sharing the same 64 N-cols:
// the 64x32 bf16 W tile is staged once per K-step (double-buffered, 1 chunk
// per thread) instead of 8x redundant L2 reads.
// ---------------------------------------------------------------------------
template <int AMODE, int SMODE>
__global__ __launch_bounds__(256) void gemm_bias_kernel(
    const void* __restrict__ Aptr, const __bf16* __restrict__ Wt,
    const float* __restrict__ bias, void* __restrict__ Out, int M, int Sper) {
  __shared__ alignas(16) __bf16 wbuf[2][64 * 32];   // 64 n-rows x 32 k, x2 buffers
  const int lane = threadIdx.x & 31;
  const int wave = threadIdx.x >> 5;
  const int hl   = lane >> 4;       // half-wave select
  const int mr   = lane & 15;
  const int koff = hl * 8;          // 16-bit operand layout K offset
  const int m0 = blockIdx.y * 128 + wave * 16;
  const int n0 = blockIdx.x * 64;
  const int sr = threadIdx.x >> 2;        // staging row 0..63
  const int sc = (threadIdx.x & 3) * 8;   // staging chunk offset (elements)
  (void)M;

  // prologue: stage first W tile
  *(v8bf*)&wbuf[0][sr * 32 + sc] = *(const v8bf*)(Wt + (size_t)(n0 + sr) * EMBED + sc);
  __syncthreads();

  v8f c[4] = {};

  for (int kk = 0; kk < EMBED; kk += 32) {
    const int cur = (kk >> 5) & 1;
    // stage next W tile into the other buffer (overlaps with WMMAs below)
    if (kk + 32 < EMBED) {
      *(v8bf*)&wbuf[cur ^ 1][sr * 32 + sc] =
          *(const v8bf*)(Wt + (size_t)(n0 + sr) * EMBED + kk + 32 + sc);
    }

    v16bf a;
    if (AMODE == 0) {
      const float* rp = (const float*)Aptr + (size_t)(m0 + mr) * EMBED + kk;
      const float4 f0 = *(const float4*)(rp + koff);
      const float4 f1 = *(const float4*)(rp + koff + 4);
      const float4 f2 = *(const float4*)(rp + 16 + koff);
      const float4 f3 = *(const float4*)(rp + 16 + koff + 4);
      a[0]=(__bf16)f0.x;  a[1]=(__bf16)f0.y;  a[2]=(__bf16)f0.z;  a[3]=(__bf16)f0.w;
      a[4]=(__bf16)f1.x;  a[5]=(__bf16)f1.y;  a[6]=(__bf16)f1.z;  a[7]=(__bf16)f1.w;
      a[8]=(__bf16)f2.x;  a[9]=(__bf16)f2.y;  a[10]=(__bf16)f2.z; a[11]=(__bf16)f2.w;
      a[12]=(__bf16)f3.x; a[13]=(__bf16)f3.y; a[14]=(__bf16)f3.z; a[15]=(__bf16)f3.w;
    } else {
      const __bf16* rp = (const __bf16*)Aptr + (size_t)(m0 + mr) * EMBED + kk;
      a = ld2x8_bf16(rp + koff, rp + 16 + koff);
    }
#pragma unroll
    for (int t = 0; t < 4; ++t) {
      const __bf16* wp = &wbuf[cur][(t * 16 + mr) * 32];
      const v16bf bm = ld2x8_bf16(wp + koff, wp + 16 + koff);
      c[t] = wmma_bf16(a, bm, c[t]);
    }
    __syncthreads();   // next tile staged & current tile consumed
  }

#pragma unroll
  for (int t = 0; t < 4; ++t) {
    const int n = n0 + t * 16 + mr;
    const float bv = bias[n];
    if (SMODE == 1) {
      // Vt epilogue: for fixed lane (fixed d), the 8 accumulator rows are 8
      // consecutive s positions -> pack into one 16B store.
      const int m = m0 + hl * 8;
      const int bb = m / Sper;
      const int s  = m - bb * Sper;
      const int hh = n >> 6;
      const int d  = n & 63;
      v8bf pk;
#pragma unroll
      for (int r = 0; r < 8; ++r) pk[r] = (__bf16)(c[t][r] + bv);
      *(v8bf*)&((__bf16*)Out)[((size_t)(bb * NHEAD + hh) * DK + d) * Sper + s] = pk;
    } else {
#pragma unroll
      for (int r = 0; r < 8; ++r) {
        const int m = m0 + r + hl * 8;      // C layout: VGPR r -> row r (+8 hi half)
        const float val = c[t][r] + bv;
        if (SMODE == 0) {
          ((__bf16*)Out)[(size_t)m * EMBED + n] = (__bf16)val;
        } else {
          ((float*)Out)[(size_t)m * EMBED + n] = val;
        }
      }
    }
  }
}

// ---------------------------------------------------------------------------
// Flash attention, block-cooperative: 8 waves share one (b,h); K/V 64-wide enc
// tiles staged in LDS (double-buffered) and reused by all 8 q-tiles.
// Row-max via shfl_xor ladder; row-sum via WMMA against all-ones B matrix.
// ---------------------------------------------------------------------------
__global__ __launch_bounds__(256) void flash_attn_kernel(
    const __bf16* __restrict__ Qp, const __bf16* __restrict__ Kp,
    const __bf16* __restrict__ Vt, __bf16* __restrict__ ctx) {
  __shared__ alignas(16) __bf16 kbuf[2][64 * 64];   // [enc 64][d 64]
  __shared__ alignas(16) __bf16 vbuf[2][64 * 64];   // [d 64][enc 64]
  __shared__ alignas(16) __bf16 pbuf[8][16 * 64];   // wave-private P tiles
  const int lane = threadIdx.x & 31;
  const int wave = threadIdx.x >> 5;
  const int hl   = lane >> 4;
  const int mr   = lane & 15;
  const int koff = hl * 8;
  const int bh = blockIdx.x >> 3;         // 64 (b,h) pairs
  const int qc = blockIdx.x & 7;          // q chunk of 128 rows
  const int b  = bh >> 4;
  const int h  = bh & 15;
  const int q0 = qc * 128 + wave * 16;    // this wave's 16 q rows
  __bf16* myP = pbuf[wave];

  const __bf16* kbase = Kp + (size_t)(b * SENC) * EMBED + h * DK;
  const __bf16* vbase = Vt + (size_t)((b * NHEAD + h) * DK) * SENC;

  // staging assignment: 512 16B chunks per tile, 2 per thread
  const int sr0 = threadIdx.x >> 2;             // rows 0..63
  const int sc0 = (threadIdx.x & 3) * 8;        // chunk offsets 0/8/16/24
  const int sc1 = sc0 + 32;                     // chunk offsets 32/40/48/56

  // Q tile as two A operands (d = 0..31, 32..63), kept in registers
  const __bf16* qrow = Qp + (size_t)(b * SDEC + q0 + mr) * EMBED + h * DK;
  const v16bf aq0 = ld2x8_bf16(qrow + koff,      qrow + 16 + koff);
  const v16bf aq1 = ld2x8_bf16(qrow + 32 + koff, qrow + 48 + koff);

  v16bf ones;
#pragma unroll
  for (int i = 0; i < 16; ++i) ones[i] = (__bf16)1.0f;

  v8f o[4] = {};
  v8f mrow, lrow;
#pragma unroll
  for (int r = 0; r < 8; ++r) { mrow[r] = -3.0e38f; lrow[r] = 0.0f; }
  const float scale = 0.125f;   // 1/sqrt(64)

  // prologue: stage tile 0
  *(v8bf*)&kbuf[0][sr0 * 64 + sc0] = *(const v8bf*)(kbase + (size_t)sr0 * EMBED + sc0);
  *(v8bf*)&kbuf[0][sr0 * 64 + sc1] = *(const v8bf*)(kbase + (size_t)sr0 * EMBED + sc1);
  *(v8bf*)&vbuf[0][sr0 * 64 + sc0] = *(const v8bf*)(vbase + (size_t)sr0 * SENC + sc0);
  *(v8bf*)&vbuf[0][sr0 * 64 + sc1] = *(const v8bf*)(vbase + (size_t)sr0 * SENC + sc1);
  __syncthreads();

  for (int e0 = 0; e0 < SENC; e0 += 64) {
    const int cur = (e0 >> 6) & 1;
    if (e0 + 64 < SENC) {   // stage next K/V tiles into the other buffer
      const int nx = cur ^ 1;
      const __bf16* kn = kbase + (size_t)(e0 + 64 + sr0) * EMBED;
      const __bf16* vn = vbase + (size_t)sr0 * SENC + e0 + 64;
      *(v8bf*)&kbuf[nx][sr0 * 64 + sc0] = *(const v8bf*)(kn + sc0);
      *(v8bf*)&kbuf[nx][sr0 * 64 + sc1] = *(const v8bf*)(kn + sc1);
      *(v8bf*)&vbuf[nx][sr0 * 64 + sc0] = *(const v8bf*)(vn + sc0);
      *(v8bf*)&vbuf[nx][sr0 * 64 + sc1] = *(const v8bf*)(vn + sc1);
    }

    // scores: four 16x16 tiles from LDS K tile (2 WMMAs each over d_k=64)
    const __bf16* kb = kbuf[cur];
    v8f s[4];
#pragma unroll
    for (int nt = 0; nt < 4; ++nt) {
      const __bf16* kr = kb + (nt * 16 + mr) * 64;
      v8f acc = {};
      acc = wmma_bf16(aq0, ld2x8_bf16(kr + koff,      kr + 16 + koff), acc);
      acc = wmma_bf16(aq1, ld2x8_bf16(kr + 32 + koff, kr + 48 + koff), acc);
      s[nt] = acc;
    }

    // online softmax: row max across 4 tiles + 16-lane half reduction
    v8f alpha;
#pragma unroll
    for (int r = 0; r < 8; ++r) {
      const float a0 = s[0][r] * scale, a1 = s[1][r] * scale;
      const float a2 = s[2][r] * scale, a3 = s[3][r] * scale;
      float mx = fmaxf(fmaxf(a0, a1), fmaxf(a2, a3));
      mx = fmaxf(mx, __shfl_xor(mx, 1, 32));
      mx = fmaxf(mx, __shfl_xor(mx, 2, 32));
      mx = fmaxf(mx, __shfl_xor(mx, 4, 32));
      mx = fmaxf(mx, __shfl_xor(mx, 8, 32));
      const float mnew = fmaxf(mrow[r], mx);
      alpha[r] = __expf(mrow[r] - mnew);
      mrow[r] = mnew;
      s[0][r] = __expf(a0 - mnew);
      s[1][r] = __expf(a1 - mnew);
      s[2][r] = __expf(a2 - mnew);
      s[3][r] = __expf(a3 - mnew);
    }

    // P: C-layout registers -> LDS (row-major 16x64) -> two A-layout operands
#pragma unroll
    for (int r = 0; r < 8; ++r) {
      const int row = r + hl * 8;
      myP[row * 64 + mr]      = (__bf16)s[0][r];
      myP[row * 64 + 16 + mr] = (__bf16)s[1][r];
      myP[row * 64 + 32 + mr] = (__bf16)s[2][r];
      myP[row * 64 + 48 + mr] = (__bf16)s[3][r];
    }
    const __bf16* pr = myP + mr * 64;
    const v16bf ap0 = ld2x8_bf16(pr + koff,      pr + 16 + koff);
    const v16bf ap1 = ld2x8_bf16(pr + 32 + koff, pr + 48 + koff);

    // row-sum of P via WMMA against all-ones B (every lane gets full row sum)
    v8f rs = {};
    rs = wmma_bf16(ap0, ones, rs);
    rs = wmma_bf16(ap1, ones, rs);
#pragma unroll
    for (int r = 0; r < 8; ++r) lrow[r] = lrow[r] * alpha[r] + rs[r];
#pragma unroll
    for (int t = 0; t < 4; ++t) o[t] *= alpha;

    // O += P @ V  from LDS V tile (4 d-column tiles x 2 k-halves)
    const __bf16* vb = vbuf[cur];
#pragma unroll
    for (int t = 0; t < 4; ++t) {
      const __bf16* vr = vb + (t * 16 + mr) * 64;
      o[t] = wmma_bf16(ap0, ld2x8_bf16(vr + koff,      vr + 16 + koff), o[t]);
      o[t] = wmma_bf16(ap1, ld2x8_bf16(vr + 32 + koff, vr + 48 + koff), o[t]);
    }

    __syncthreads();   // next tiles staged & current tiles fully consumed
  }

  v8f inv;
#pragma unroll
  for (int r = 0; r < 8; ++r) inv[r] = 1.0f / lrow[r];
#pragma unroll
  for (int t = 0; t < 4; ++t) {
#pragma unroll
    for (int r = 0; r < 8; ++r) {
      ctx[(size_t)(b * SDEC + q0 + r + hl * 8) * EMBED + h * DK + t * 16 + mr] =
          (__bf16)(o[t][r] * inv[r]);
    }
  }
}

// ---------------------------------------------------------------------------
extern "C" void kernel_launch(void* const* d_in, const int* in_sizes, int n_in,
                              void* d_out, int out_size, void* d_ws, size_t ws_size,
                              hipStream_t stream) {
  (void)in_sizes; (void)n_in; (void)out_size; (void)ws_size;
  const float* Qd = (const float*)d_in[0];
  const float* Ke = (const float*)d_in[1];
  const float* Ve = (const float*)d_in[2];
  const float* Wq = (const float*)d_in[3];
  const float* bq = (const float*)d_in[4];
  const float* Wk = (const float*)d_in[5];
  const float* bk = (const float*)d_in[6];
  const float* Wv = (const float*)d_in[7];
  const float* bv = (const float*)d_in[8];
  const float* Wo = (const float*)d_in[9];
  const float* bo = (const float*)d_in[10];
  float* out = (float*)d_out;

  // Workspace carve (bf16 intermediates, ~56 MB total)
  uint8_t* ws = (uint8_t*)d_ws;
  size_t off = 0;
  auto carve = [&](size_t bytes) -> void* {
    void* p = ws + off;
    off += (bytes + 255) & ~(size_t)255;
    return p;
  };
  __bf16* WqT = (__bf16*)carve((size_t)EMBED * EMBED * 2);
  __bf16* WkT = (__bf16*)carve((size_t)EMBED * EMBED * 2);
  __bf16* WvT = (__bf16*)carve((size_t)EMBED * EMBED * 2);
  __bf16* WoT = (__bf16*)carve((size_t)EMBED * EMBED * 2);
  __bf16* Qp  = (__bf16*)carve((size_t)BATCH * SDEC * EMBED * 2);
  __bf16* Kp  = (__bf16*)carve((size_t)BATCH * SENC * EMBED * 2);
  __bf16* Vtr = (__bf16*)carve((size_t)BATCH * SENC * EMBED * 2);
  __bf16* ctx = (__bf16*)carve((size_t)BATCH * SDEC * EMBED * 2);

  const dim3 blk(256);

  // 1) weights -> transposed bf16
  wt_bf16_kernel<<<4096, blk, 0, stream>>>(Wq, WqT);
  wt_bf16_kernel<<<4096, blk, 0, stream>>>(Wk, WkT);
  wt_bf16_kernel<<<4096, blk, 0, stream>>>(Wv, WvT);
  wt_bf16_kernel<<<4096, blk, 0, stream>>>(Wo, WoT);

  // 2-4) projections (f32 A converted on load, bf16 WMMA, f32 accum)
  gemm_bias_kernel<0, 0><<<dim3(16, 32), blk, 0, stream>>>(Qd, WqT, bq, Qp,
                                                           BATCH * SDEC, SDEC);
  gemm_bias_kernel<0, 0><<<dim3(16, 64), blk, 0, stream>>>(Ke, WkT, bk, Kp,
                                                           BATCH * SENC, SENC);
  gemm_bias_kernel<0, 1><<<dim3(16, 64), blk, 0, stream>>>(Ve, WvT, bv, Vtr,
                                                           BATCH * SENC, SENC);

  // 5) flash attention: 64 (b,h) pairs x 8 q-chunks, 8 waves per block
  flash_attn_kernel<<<512, blk, 0, stream>>>(Qp, Kp, Vtr, ctx);

  // 6) output projection -> f32 d_out
  gemm_bias_kernel<1, 2><<<dim3(16, 32), blk, 0, stream>>>(ctx, WoT, bo, out,
                                                           BATCH * SDEC, SDEC);
}
